// GameNet_1924145349063
// MI455X (gfx1250) — matrix operand order repaired and checked by reading, not compiled, not run on hardware
//
#include <hip/hip_runtime.h>
#include <hip/hip_bf16.h>

typedef __attribute__((ext_vector_type(16))) _Float16 v16h;
typedef __attribute__((ext_vector_type(8)))  _Float16 v8h;
typedef __attribute__((ext_vector_type(4)))  _Float16 v4h;
typedef __attribute__((ext_vector_type(8)))  float    v8f;

#define DNN   256
#define NTOT  262144      // 2048 graphs * 128 nodes
#define NGPH  2048
#define NPG   128
#define EPERG 1024        // 128 nodes * deg 8
#define ETOT  (NTOT * 8)
#define BSTR  264         // padded LDS row stride (halves): 528B = 132 dwords -> bank-spread

// ---------------------------------------------------------------------------
// fragment helpers (contiguous k order for A and B: lane half h holds k = k0+16h..k0+16h+15)
// ---------------------------------------------------------------------------
__device__ __forceinline__ v16h ld_afrag(const _Float16* row, int k0, int kA) {
  v8h lo = *(const v8h*)(row + k0 + 2 * kA);
  v8h hi = *(const v8h*)(row + k0 + 8 + 2 * kA);
  return __builtin_shufflevector(lo, hi, 0, 1, 2, 3, 4, 5, 6, 7,
                                 8, 9, 10, 11, 12, 13, 14, 15);
}
__device__ __forceinline__ v16h ld_bfrag(const _Float16* p) {  // 16 contiguous halves
  v8h lo = *(const v8h*)(p);
  v8h hi = *(const v8h*)(p + 8);
  return __builtin_shufflevector(lo, hi, 0, 1, 2, 3, 4, 5, 6, 7,
                                 8, 9, 10, 11, 12, 13, 14, 15);
}

// CDNA5 async global->LDS copy (ASYNCcnt) + wait
__device__ __forceinline__ void async_g2l_b128(unsigned ldsAddr, const void* gaddr) {
  asm volatile("global_load_async_to_lds_b128 %0, %1, off"
               :: "v"(ldsAddr), "v"((unsigned long long)gaddr) : "memory");
}
__device__ __forceinline__ void wait_async0() {
  asm volatile("s_wait_asynccnt 0x0" ::: "memory");
}

// ---------------------------------------------------------------------------
// Weight transpose + f32->f16 convert: WT[n*256+k] = (f16)W[k*256+n]
// ---------------------------------------------------------------------------
__global__ void k_wconvT(const float* __restrict__ W, _Float16* __restrict__ WT) {
  int k = blockIdx.x;
  int n = threadIdx.x;
  WT[(size_t)n * DNN + k] = (_Float16)W[(size_t)k * DNN + n];
}

__global__ void k_padcopy(const float* __restrict__ src, _Float16* __restrict__ dst,
                          int validElems, int totalElems) {
  int i = blockIdx.x * 256 + threadIdx.x;
  if (i < totalElems)
    dst[i] = (i < validElems) ? (_Float16)src[i] : (_Float16)0.f;
}

// ---------------------------------------------------------------------------
// Input layer: h = relu(x @ Win + b), x[NTOT,20], Win[20,256]
// ---------------------------------------------------------------------------
__global__ void k_input(const float* __restrict__ x, const float* __restrict__ W,
                        const float* __restrict__ b, _Float16* __restrict__ h) {
  __shared__ float Ws[20 * 256];
  __shared__ float xs[32 * 20];
  int tid = threadIdx.x;
  for (int i = tid; i < 20 * 256; i += 256) Ws[i] = W[i];
  int rb = blockIdx.x * 32;
  for (int i = tid; i < 32 * 20; i += 256) xs[i] = x[(size_t)rb * 20 + i];
  __syncthreads();
  float bias = b[tid];
  for (int r = 0; r < 32; ++r) {
    float acc = bias;
#pragma unroll
    for (int k = 0; k < 20; ++k) acc += xs[r * 20 + k] * Ws[k * 256 + tid];
    h[(size_t)(rb + r) * DNN + tid] = (_Float16)fmaxf(acc, 0.f);
  }
}

// ---------------------------------------------------------------------------
// SAGE mean aggregation: one block per graph, LDS f32 accumulation (ds_add_f32)
// ---------------------------------------------------------------------------
__global__ void k_aggregate(const _Float16* __restrict__ h, const int* __restrict__ ei,
                            _Float16* __restrict__ mean) {
  __shared__ float acc[NPG * 128];
  __shared__ float cnt[NPG];
  int g = blockIdx.x;
  int tid = threadIdx.x;
  int lane = tid & 31;
  int wave = tid >> 5;
  int ebase = g * EPERG;
  int nbase = g * NPG;

  if (tid < NPG) cnt[tid] = 0.f;
  __syncthreads();
  for (int e = tid; e < EPERG; e += 256) {
    int d = ei[ETOT + ebase + e] & (NPG - 1);
    atomicAdd(&cnt[d], 1.f);
  }
  __syncthreads();

  for (int half = 0; half < 2; ++half) {
    for (int i = tid; i < NPG * 128; i += 256) acc[i] = 0.f;
    __syncthreads();
    for (int e = wave; e < EPERG; e += 8) {
      int s = ei[ebase + e];
      int d = ei[ETOT + ebase + e] & (NPG - 1);
      v4h v = *(const v4h*)(h + (size_t)s * DNN + half * 128 + lane * 4);
#pragma unroll
      for (int j = 0; j < 4; ++j)
        atomicAdd(&acc[d * 128 + lane * 4 + j], (float)v[j]);
    }
    __syncthreads();
    for (int i = tid; i < NPG * 128; i += 256) {
      int node = i >> 7, f = i & 127;
      float m = acc[i] / fmaxf(cnt[node], 1.f);
      mean[(size_t)(nbase + node) * DNN + half * 128 + f] = (_Float16)m;
    }
    __syncthreads();
  }
}

// ---------------------------------------------------------------------------
// WMMA GEMM with async-staged B in LDS.
// C[M,256] = ep( A@B1 (+A2@B2) + bias ); BT pre-transposed f16 [N,K].
// Block = 8 waves = 2 rowTiles x 4 col groups. Whole B (and B2) DMA'd to LDS
// once per block via global_load_async_to_lds_b128 (ASYNCcnt path).
// mode: 0 = none, 1 = relu, 2 = resid + relu
// ---------------------------------------------------------------------------
template <bool DUAL>
__global__ void k_gemm256(const _Float16* __restrict__ Aa, const _Float16* __restrict__ A2,
                          const _Float16* __restrict__ BT, const _Float16* __restrict__ B2T,
                          const float* __restrict__ bias, const _Float16* __restrict__ resid,
                          _Float16* __restrict__ C, int M, int mode) {
  __shared__ __align__(16) _Float16 lB[(DUAL ? 2 : 1) * 256 * BSTR];
  int tid = threadIdx.x;
  int wave = (blockIdx.x * blockDim.x + tid) >> 5;
  int lane = tid & 31;
  int rowTile = wave >> 2;
  int cg = wave & 3;
  int l15 = lane & 15;
  int kA = (lane < 16) ? 0 : 8;
  int kB = (lane < 16) ? 0 : 16;

  // ---- async DMA: stage B (and B2) rows into padded LDS ------------------
  unsigned ldsBase = (unsigned)(size_t)&lB[0];
#pragma unroll 4
  for (int j = 0; j < 32; ++j) {
    int G = j * 256 + tid;             // 8192 x 16B granules = 128KB
    int row = G >> 5, chunk = G & 31;
    unsigned dst = ldsBase + (unsigned)(row * (BSTR * 2) + chunk * 16);
    async_g2l_b128(dst, (const char*)BT + row * 512 + chunk * 16);
    if (DUAL)
      async_g2l_b128(dst + 256u * (BSTR * 2),
                     (const char*)B2T + row * 512 + chunk * 16);
  }
  wait_async0();
  __syncthreads();

  bool valid = (rowTile * 16) < M;
  int m = (valid ? rowTile * 16 : 0) + l15;
  const _Float16* arow = Aa + (size_t)m * DNN;
  const _Float16* a2row = DUAL ? (A2 + (size_t)m * DNN) : (const _Float16*)0;

  if (valid) {
    v8f acc[4] = {v8f{}, v8f{}, v8f{}, v8f{}};
    for (int k0 = 0; k0 < DNN; k0 += 32) {
      v16h a = ld_afrag(arow, k0, kA);
#pragma unroll
      for (int t = 0; t < 4; ++t) {
        v16h b = ld_bfrag(&lB[(cg * 64 + t * 16 + l15) * BSTR + k0 + kB]);
        acc[t] = __builtin_amdgcn_wmma_f32_16x16x32_f16(false, a, false, b,
                                                        (short)0, acc[t], false, false);
      }
      if (DUAL) {
        v16h a2 = ld_afrag(a2row, k0, kA);
#pragma unroll
        for (int t = 0; t < 4; ++t) {
          v16h b2 = ld_bfrag(&lB[256 * BSTR + (cg * 64 + t * 16 + l15) * BSTR + k0 + kB]);
          acc[t] = __builtin_amdgcn_wmma_f32_16x16x32_f16(false, a2, false, b2,
                                                          (short)0, acc[t], false, false);
        }
      }
    }
#pragma unroll
    for (int t = 0; t < 4; ++t) {
      int n = cg * 64 + t * 16 + l15;
      float bb = bias ? bias[n] : 0.f;
#pragma unroll
      for (int r = 0; r < 8; ++r) {
        int row = rowTile * 16 + ((lane < 16) ? r : 8 + r);
        float v = acc[t][r] + bb;
        if (mode >= 1) v = fmaxf(v, 0.f);
        if (mode == 2) v += (float)resid[(size_t)row * DNN + n];
        C[(size_t)row * DNN + n] = (_Float16)v;
      }
    }
  }
}

// ---------------------------------------------------------------------------
// MAB attention core, one block per graph, 4 heads of dh=64.
// scores = softmax(Qh Kh^T / 16) ; out = Qp + scores @ Vh
// Padded LDS strides to avoid 64-bank conflicts.
// ---------------------------------------------------------------------------
#define SCP 132   // f32 score row stride
#define PRP 136   // f16 prob row stride
#define VTP 136   // f16 V^T row stride
__global__ void k_attn(const _Float16* __restrict__ Qp, const _Float16* __restrict__ Kp,
                       const _Float16* __restrict__ Vp, _Float16* __restrict__ outp,
                       int nq, int nqPad, int nk, int nkPad, int qShared) {
  __shared__ __align__(16) float    sc[80 * SCP];
  __shared__ __align__(16) _Float16 pr[80 * PRP];
  __shared__ __align__(16) _Float16 vt[64 * VTP];
  int g = blockIdx.x;
  const _Float16* Q = qShared ? Qp : (Qp + (size_t)g * nq * DNN);
  const _Float16* K = Kp + (size_t)g * nk * DNN;
  const _Float16* V = Vp + (size_t)g * nk * DNN;
  _Float16* O = outp + (size_t)g * nq * DNN;

  int tid = threadIdx.x, wave = tid >> 5, lane = tid & 31, l15 = lane & 15;
  int kA = (lane < 16) ? 0 : 8;
  int kB = (lane < 16) ? 0 : 16;
  int nQT = nqPad >> 4, nKT = nkPad >> 4;

  for (int h = 0; h < 4; ++h) {
    int hoff = h * 64;
    // ---- scores = Q K^T / sqrt(256) -------------------------------------
    for (int t = wave; t < nQT * nKT; t += 8) {
      int qt = t / nKT, kt = t % nKT;
      v8f a8 = {};
      for (int k0 = 0; k0 < 64; k0 += 32) {
        v16h a = ld_afrag(Q + (size_t)(qt * 16 + l15) * DNN + hoff, k0, kA);
        v16h b = ld_bfrag(K + (size_t)(kt * 16 + l15) * DNN + hoff + k0 + kB);
        a8 = __builtin_amdgcn_wmma_f32_16x16x32_f16(false, a, false, b,
                                                    (short)0, a8, false, false);
      }
#pragma unroll
      for (int r = 0; r < 8; ++r) {
        int row = qt * 16 + ((lane < 16) ? r : 8 + r);
        sc[row * SCP + kt * 16 + l15] = a8[r] * 0.0625f;
      }
    }
    __syncthreads();
    // ---- softmax rows ----------------------------------------------------
    if (tid < nqPad) {
      float mx = -3.0e38f;
      for (int k = 0; k < nk; ++k) mx = fmaxf(mx, sc[tid * SCP + k]);
      float s = 0.f;
      for (int k = 0; k < nk; ++k) {
        float e = __expf(sc[tid * SCP + k] - mx);
        sc[tid * SCP + k] = e; s += e;
      }
      float inv = 1.f / s;
      for (int k = 0; k < nkPad; ++k)
        pr[tid * PRP + k] = (k < nk) ? (_Float16)(sc[tid * SCP + k] * inv) : (_Float16)0.f;
    }
    // ---- stage V^T for this head ----------------------------------------
    for (int i = tid; i < 64 * nkPad; i += 256) {
      int d = i / nkPad, key = i % nkPad;
      vt[d * VTP + key] = (key < nk) ? V[(size_t)key * DNN + hoff + d] : (_Float16)0.f;
    }
    __syncthreads();
    // ---- O = P @ V ; out = Qp + O ---------------------------------------
    for (int t = wave; t < nQT * 4; t += 8) {
      int qt = t >> 2, ct = t & 3;
      v8f a8 = {};
      for (int k0 = 0; k0 < nkPad; k0 += 32) {
        v16h a = ld_afrag(pr + (qt * 16 + l15) * PRP, k0, kA);
        v16h b = ld_bfrag(vt + (ct * 16 + l15) * VTP + k0 + kB);
        a8 = __builtin_amdgcn_wmma_f32_16x16x32_f16(false, a, false, b,
                                                    (short)0, a8, false, false);
      }
#pragma unroll
      for (int r = 0; r < 8; ++r) {
        int row = qt * 16 + ((lane < 16) ? r : 8 + r);
        if (row < nq) {
          int n = hoff + ct * 16 + l15;
          O[(size_t)row * DNN + n] = (_Float16)((float)Q[(size_t)row * DNN + n] + a8[r]);
        }
      }
    }
    __syncthreads();
  }
}

// ---------------------------------------------------------------------------
// Final lin2: y[b] = X[b,:] . W + bias
// ---------------------------------------------------------------------------
__global__ void k_lin2(const _Float16* __restrict__ X, const float* __restrict__ W,
                       const float* __restrict__ b, float* __restrict__ y, int Bn) {
  int row = blockIdx.x * 8 + (threadIdx.x >> 5);
  int lane = threadIdx.x & 31;
  if (row >= Bn) return;
  float s = 0.f;
  for (int k = lane; k < DNN; k += 32) s += (float)X[(size_t)row * DNN + k] * W[k];
#pragma unroll
  for (int off = 16; off > 0; off >>= 1) s += __shfl_xor(s, off, 32);
  if (lane == 0) y[row] = s + b[0];
}

// ---------------------------------------------------------------------------
extern "C" void kernel_launch(void* const* d_in, const int* in_sizes, int n_in,
                              void* d_out, int out_size, void* d_ws, size_t ws_size,
                              hipStream_t stream) {
  (void)in_sizes; (void)n_in; (void)out_size; (void)ws_size;
  const float* x   = (const float*)d_in[0];
  const int*   ei  = (const int*)d_in[1];
  const float* Win = (const float*)d_in[4];
  const float* bin = (const float*)d_in[5];
  // convs: Wl=6+3i, bl=7+3i, Wr=8+3i ; lin1 W=15 b=16 ; S1=17
  // mab1 = 18..25 ; mab2 = 26..33 ; S3=34 ; mab3 = 35..42 ; lin2 W=43 b=44

  size_t off = 0;
  char* ws = (char*)d_ws;
  auto carve = [&](size_t bytes) -> void* {
    void* p = ws + off;
    off += (bytes + 255) & ~(size_t)255;
    return p;
  };
  _Float16* wt[19];
  for (int i = 0; i < 19; ++i) wt[i] = (_Float16*)carve((size_t)DNN * DNN * 2);
  _Float16* s1p = (_Float16*)carve(80 * 256 * 2);
  _Float16* qp1 = (_Float16*)carve(80 * 256 * 2);
  _Float16* s3p = (_Float16*)carve(16 * 256 * 2);
  _Float16* qp3 = (_Float16*)carve(16 * 256 * 2);
  _Float16* big[4];
  for (int i = 0; i < 4; ++i) big[i] = (_Float16*)carve((size_t)NTOT * DNN * 2 + 16384);

  const int wsrc[19] = {6, 9, 12,  8, 11, 14,  15,
                        18, 20, 22, 24,  26, 28, 30, 32,  35, 37, 39, 41};
  for (int i = 0; i < 19; ++i)
    k_wconvT<<<256, 256, 0, stream>>>((const float*)d_in[wsrc[i]], wt[i]);
  k_padcopy<<<(80 * 256 + 255) / 256, 256, 0, stream>>>((const float*)d_in[17], s1p,
                                                        75 * 256, 80 * 256);
  k_padcopy<<<(16 * 256 + 255) / 256, 256, 0, stream>>>((const float*)d_in[34], s3p,
                                                        1 * 256, 16 * 256);

  auto gemm1 = [&](const _Float16* A, const _Float16* BT, int biasIdx,
                   const _Float16* resid, _Float16* C, int M, int mode) {
    int blocks = ((M / 16) * 4 + 7) / 8;
    k_gemm256<false><<<blocks, 256, 0, stream>>>(A, 0, BT, 0,
        biasIdx >= 0 ? (const float*)d_in[biasIdx] : (const float*)0, resid, C, M, mode);
  };
  auto gemm2 = [&](const _Float16* A, const _Float16* A2, const _Float16* BT,
                   const _Float16* B2T, int biasIdx, _Float16* C, int M, int mode) {
    int blocks = ((M / 16) * 4 + 7) / 8;
    k_gemm256<true><<<blocks, 256, 0, stream>>>(A, A2, BT, B2T,
        biasIdx >= 0 ? (const float*)d_in[biasIdx] : (const float*)0, 0, C, M, mode);
  };

  // ---- input layer -------------------------------------------------------
  k_input<<<NTOT / 32, 256, 0, stream>>>(x, Win, bin, big[0]);

  // ---- 3 SAGE convs: h' = relu(mean@Wl + bl + h@Wr) ----------------------
  _Float16* cur = big[0];
  _Float16* outb[3] = {big[2], big[0], big[2]};
  for (int i = 0; i < 3; ++i) {
    k_aggregate<<<NGPH, 256, 0, stream>>>(cur, ei, big[1]);
    gemm2(big[1], cur, wt[i], wt[3 + i], 7 + 3 * i, outb[i], NTOT, 1);
    cur = outb[i];
  }
  // cur == big[2]

  // ---- lin1 (no relu) ----------------------------------------------------
  gemm1(cur, wt[6], 16, 0, big[0], NTOT, 0);

  // ---- mab1: PMA with 75 seeds ------------------------------------------
  gemm1(s1p, wt[7], 19, 0, qp1, 80, 0);
  gemm1(big[0], wt[8], 21, 0, big[1], NTOT, 0);                  // Kp1
  gemm1(big[0], wt[9], 23, 0, big[2], NTOT, 0);                  // Vp1
  k_attn<<<NGPH, 256, 0, stream>>>(qp1, big[1], big[2], big[3],
                                   75, 80, 128, 128, 1);
  const int M75 = NGPH * 75;
  gemm1(big[3], wt[10], 25, big[3], big[0], M75, 2);             // X1 -> big0

  // ---- mab2: SAB over 75 tokens -----------------------------------------
  gemm1(big[0], wt[11], 27, 0, big[1], M75, 0);                  // Qp2
  gemm1(big[0], wt[12], 29, 0, big[2], M75, 0);                  // Kp2
  gemm1(big[0], wt[13], 31, 0, big[3], M75, 0);                  // Vp2
  k_attn<<<NGPH, 256, 0, stream>>>(big[1], big[2], big[3], big[0],
                                   75, 80, 75, 96, 0);
  gemm1(big[0], wt[14], 33, big[0], big[1], M75, 2);             // X2 -> big1

  // ---- mab3: PMA with 1 seed --------------------------------------------
  gemm1(s3p, wt[15], 36, 0, qp3, 16, 0);
  gemm1(big[1], wt[16], 38, 0, big[2], M75, 0);                  // Kp3
  gemm1(big[1], wt[17], 40, 0, big[3], M75, 0);                  // Vp3
  k_attn<<<NGPH, 256, 0, stream>>>(qp3, big[2], big[3], big[0],
                                   1, 16, 75, 96, 1);            // pooled -> big0
  gemm1(big[0], wt[18], 42, big[0], big[1], NGPH, 2);            // pooled2 -> big1

  // ---- lin2 -> d_out [B] f32 --------------------------------------------
  k_lin2<<<NGPH / 8, 256, 0, stream>>>(big[1], (const float*)d_in[43],
                                       (const float*)d_in[44], (float*)d_out, NGPH);
}
